// DiscreteGNNDenoiser_8169027797463
// MI455X (gfx1250) — matrix operand order repaired and checked by reading, not compiled, not run on hardware
//
#include <hip/hip_runtime.h>
#include <hip/hip_bf16.h>

typedef __attribute__((ext_vector_type(16))) _Float16 v16h;
typedef __attribute__((ext_vector_type(2)))  _Float16 v2h;
typedef __attribute__((ext_vector_type(8)))  float    v8f;

#define WAVES 8

#if defined(__has_builtin)
#if __has_builtin(__builtin_amdgcn_global_load_async_to_lds_b64) && \
    __has_builtin(__builtin_amdgcn_s_wait_asynccnt)
#define HAVE_ASYNC 1
#endif
#endif
#ifndef HAVE_ASYNC
#define HAVE_ASYNC 0
#endif

#if HAVE_ASYNC
typedef __attribute__((ext_vector_type(2))) int v2i;
typedef v2i __attribute__((address_space(1))) v2i_g;   // global (device) AS
typedef v2i __attribute__((address_space(3))) v2i_l;   // LDS (shared) AS
#endif

__device__ __forceinline__ v8f wmma16(v16h a, v16h b, v8f c) {
  return __builtin_amdgcn_wmma_f32_16x16x32_f16(false, a, false, b, (short)0, c, false, false);
}

// Inverse of this file's 16x32 f16 fragment layout (contiguous k = 16*hsel + i): element (row, kk) of a
// 32-wide K window lives at fragment lane hsel*16+row, vector slot i.
// Returns half-index within the 512-half window region.
__device__ __forceinline__ int frag_pos(int row, int kk) {
  int lanei = ((kk >> 4) << 4) | row;
  int i = kk & 15;
  return lanei * 16 + i;
}

// Fragment loads: one v16h per lane (lowers to 2x ds_load_b128).
__device__ __forceinline__ v16h load_frag(const _Float16* base, int region, int lane) {
  return *(const v16h*)(base + region * 512 + lane * 16);
}

__global__ void zero_kernel(float* __restrict__ p, int n) {
  int i = blockIdx.x * blockDim.x + threadIdx.x;
  if (i < n) p[i] = 0.f;
}

__global__ void count_kernel(const int* __restrict__ ei, float* __restrict__ cnt, int En) {
  int e = blockIdx.x * blockDim.x + threadIdx.x;
  if (e < En) atomicAdd(&cnt[ei[En + e]], 1.f);
}

// Per-node prep: h0 = [x, active_sites, 0-pad], time embedding, cond MLP (4->32->32).
__global__ void prep_kernel(const float* __restrict__ x, const float* __restrict__ act,
                            const int* __restrict__ bidx, const float* __restrict__ timev,
                            const float* __restrict__ y,
                            const float* __restrict__ cw1, const float* __restrict__ cb1,
                            const float* __restrict__ cw2, const float* __restrict__ cb2,
                            float* __restrict__ h, float* __restrict__ temb,
                            float* __restrict__ cemb, int Nn) {
  int n = blockIdx.x * blockDim.x + threadIdx.x;
  if (n >= Nn) return;
#pragma unroll
  for (int k = 0; k < 10; ++k) h[(long)n * 32 + k] = x[(long)n * 10 + k];
  h[(long)n * 32 + 10] = act[(long)n * 2 + 0];
  h[(long)n * 32 + 11] = act[(long)n * 2 + 1];
#pragma unroll
  for (int k = 12; k < 32; ++k) h[(long)n * 32 + k] = 0.f;
  int b = bidx[n];
  float t = timev[b];
  float ang = 3.14159265358979323846f * t;
  temb[(long)n * 2 + 0] = __cosf(ang);
  temb[(long)n * 2 + 1] = __sinf(ang);
  float yy[4];
#pragma unroll
  for (int i = 0; i < 4; ++i) yy[i] = y[(long)b * 4 + i];
  float h1[32];
#pragma unroll 4
  for (int o = 0; o < 32; ++o) {
    float a = cb1[o];
#pragma unroll
    for (int i = 0; i < 4; ++i) a += yy[i] * cw1[i * 32 + o];
    h1[o] = a > 0.f ? a : 0.f;
  }
#pragma unroll 4
  for (int o = 0; o < 32; ++o) {
    float a = cb2[o];
    for (int i = 0; i < 32; ++i) a += h1[i] * cw2[i * 32 + o];
    cemb[(long)n * 32 + o] = a;
  }
}

// psi: per 16-edge tile, m = relu([h[src],ea] @ W1 + b1) @ W2 + b2 ; scatter-add to s[dst].
__global__ __launch_bounds__(256) void edge_kernel(
    const float* __restrict__ h, int in_dim,
    const int* __restrict__ ei, const float* __restrict__ ea,
    const float* __restrict__ w1, const float* __restrict__ b1,
    const float* __restrict__ w2, const float* __restrict__ b2,
    float* __restrict__ s, int En) {
  // all weight / tile storage is fragment-ordered: [region][lane][16]
  __shared__ __align__(32) _Float16 Wl1[2 * 2 * 512];   // [wnd][nb]
  __shared__ __align__(32) _Float16 Wl2[2 * 512];       // [nb]
  __shared__ __align__(32) _Float16 stage[WAVES][2 * 512];
#if HAVE_ASYNC
  __shared__ __align__(16) unsigned char eaLds[WAVES][16 * 48];
#endif
  int tid = threadIdx.x, lane = tid & 31, wave = tid >> 5;
  int K1 = in_dim + 12;
  int Kp1 = (K1 + 31) & ~31;
  int nw = Kp1 >> 5;
  int tile = blockIdx.x * WAVES + wave;
  int e0 = tile * 16;

#if HAVE_ASYNC
  { // async-copy this wave's 16x12 f32 edge_attr tile (768 contiguous bytes)
    const unsigned char* g = (const unsigned char*)ea + (size_t)e0 * 48;
    unsigned char* l = &eaLds[wave][0];
#pragma unroll
    for (int c = 0; c < 3; ++c) {
      int boff = c * 256 + lane * 8;            // 8B chunks; 48B rows => row-aligned
      if (e0 + boff / 48 < En)
        __builtin_amdgcn_global_load_async_to_lds_b64(
            (v2i_g*)(g + boff), (v2i_l*)(l + boff), 0, 0);
    }
  }
#endif

  // stage weights in fragment order (B layout: lane = (k>>4 within wnd)*16+col, elem i = k&15)
  for (int idx = tid; idx < nw * 2 * 512; idx += 256) {
    int wnd = idx >> 10, nb = (idx >> 9) & 1, p = idx & 511;
    int ln = p >> 4, i = p & 15;
    int k = wnd * 32 + ((ln >> 4) << 4) + i;
    int n = nb * 16 + (ln & 15);
    Wl1[idx] = (k < K1) ? (_Float16)w1[k * 32 + n] : (_Float16)0.f;
  }
  for (int idx = tid; idx < 1024; idx += 256) {
    int nb = (idx >> 9) & 1, p = idx & 511;
    int ln = p >> 4, i = p & 15;
    int k = ((ln >> 4) << 4) + i;
    int n = nb * 16 + (ln & 15);
    Wl2[idx] = (_Float16)w2[k * 32 + n];
  }

  // stage A tile in fragment order, paired 4B writes
  _Float16* sta = &stage[wave][0];
  {
    int row = lane >> 1, part = lane & 1;
    int e = e0 + row;
    bool v = e < En;
    int srcn = v ? ei[e] : 0;
#if HAVE_ASYNC
    __builtin_amdgcn_s_wait_asynccnt(0);
    const float* earow = (const float*)&eaLds[wave][row * 48];
#endif
    auto getf = [&](int k) -> float {
      if (!v) return 0.f;
      if (k < in_dim) return h[(long)srcn * 32 + k];
      if (k < K1) {
#if HAVE_ASYNC
        return earow[k - in_dim];
#else
        return ea[(long)e * 12 + (k - in_dim)];
#endif
      }
      return 0.f;
    };
    int halfK = Kp1 >> 1;
    for (int kk = 0; kk < halfK; kk += 2) {
      int k = part * halfK + kk;
      v2h pr;
      pr[0] = (_Float16)getf(k);
      pr[1] = (_Float16)getf(k + 1);
      *(v2h*)(sta + (k >> 5) * 512 + frag_pos(row, k & 31)) = pr;
    }
  }
  __syncthreads();

  // GEMM1: (16 x Kp1) @ (Kp1 x 32)
  v8f acc0 = {}, acc1 = {};
  for (int wnd = 0; wnd < nw; ++wnd) {
    v16h a  = load_frag(sta, wnd, lane);
    v16h bf0 = load_frag(Wl1, wnd * 2 + 0, lane);
    v16h bf1 = load_frag(Wl1, wnd * 2 + 1, lane);
    acc0 = wmma16(a, bf0, acc0);
    acc1 = wmma16(a, bf1, acc1);
  }
  __syncthreads();
  { // relu + bias -> hidden tile, written directly in fragment order (window 0)
    int col = lane & 15, hsel = lane >> 4;
#pragma unroll
    for (int j = 0; j < 8; ++j) {
      int M = j + hsel * 8;
      float v0 = acc0[j] + b1[col];
      float v1 = acc1[j] + b1[16 + col];
      sta[frag_pos(M, col)]      = (_Float16)(v0 > 0.f ? v0 : 0.f);
      sta[frag_pos(M, col + 16)] = (_Float16)(v1 > 0.f ? v1 : 0.f);
    }
  }
  __syncthreads();

  // GEMM2: (16x32) @ (32x32)
  v8f m0 = {}, m1 = {};
  {
    v16h a  = load_frag(sta, 0, lane);
    v16h bf0 = load_frag(Wl2, 0, lane);
    v16h bf1 = load_frag(Wl2, 1, lane);
    m0 = wmma16(a, bf0, m0);
    m1 = wmma16(a, bf1, m1);
  }
  { // scatter-add
    int col = lane & 15, hsel = lane >> 4;
#pragma unroll
    for (int j = 0; j < 8; ++j) {
      int M = j + hsel * 8;
      int e = e0 + M;
      if (e < En) {
        int d = ei[En + e];
        atomicAdd(&s[(long)d * 32 + col],      m0[j] + b2[col]);
        atomicAdd(&s[(long)d * 32 + 16 + col], m1[j] + b2[16 + col]);
      }
    }
  }
}

// phi + FiLM per 16-node tile: hnew = phi([h,agg,temb]); out = gamma(cond)*hnew + beta(cond)
__global__ __launch_bounds__(256) void node_kernel(
    const float* __restrict__ h, int in_dim, int out_dim,
    const float* __restrict__ s, const float* __restrict__ cnt,
    const float* __restrict__ temb, const float* __restrict__ cemb,
    const float* __restrict__ pw1, const float* __restrict__ pb1,
    const float* __restrict__ pw2, const float* __restrict__ pb2,
    const float* __restrict__ gw1, const float* __restrict__ gb1,
    const float* __restrict__ gw2, const float* __restrict__ gb2,
    const float* __restrict__ bw1, const float* __restrict__ bb1,
    const float* __restrict__ bw2, const float* __restrict__ bb2,
    float* __restrict__ out, int out_stride, int Nn) {
  __shared__ __align__(32) _Float16 Pw1[3 * 4 * 512];   // [wnd][nb]
  __shared__ __align__(32) _Float16 Pw2[2 * 2 * 512];   // [wnd][nb]
  __shared__ __align__(32) _Float16 Gw1[2 * 512];
  __shared__ __align__(32) _Float16 Gw2[2 * 512];
  __shared__ __align__(32) _Float16 Bw1[2 * 512];
  __shared__ __align__(32) _Float16 Bw2[2 * 512];
  __shared__ __align__(32) _Float16 stage[WAVES][3 * 512];
  int tid = threadIdx.x, lane = tid & 31, wave = tid >> 5;
  int K2 = in_dim + 34;
  int Kp2 = (K2 + 31) & ~31;
  int nw = Kp2 >> 5;
  int H2 = 32 + out_dim;
  int noutb = (out_dim + 15) >> 4;

  for (int idx = tid; idx < nw * 4 * 512; idx += 256) {
    int wnd = idx >> 11, nb = (idx >> 9) & 3, p = idx & 511;
    int ln = p >> 4, i = p & 15;
    int k = wnd * 32 + ((ln >> 4) << 4) + i;
    int n = nb * 16 + (ln & 15);
    Pw1[idx] = (k < K2 && n < H2) ? (_Float16)pw1[k * H2 + n] : (_Float16)0.f;
  }
  for (int idx = tid; idx < 2048; idx += 256) {
    int wnd = (idx >> 10) & 1, nb = (idx >> 9) & 1, p = idx & 511;
    int ln = p >> 4, i = p & 15;
    int k = wnd * 32 + ((ln >> 4) << 4) + i;
    int n = nb * 16 + (ln & 15);
    Pw2[idx] = (k < H2 && n < out_dim) ? (_Float16)pw2[k * out_dim + n] : (_Float16)0.f;
  }
  for (int idx = tid; idx < 1024; idx += 256) {
    int nb = (idx >> 9) & 1, p = idx & 511;
    int ln = p >> 4, i = p & 15;
    int k = ((ln >> 4) << 4) + i;
    int n = nb * 16 + (ln & 15);
    Gw1[idx] = (_Float16)gw1[k * 32 + n];
    Bw1[idx] = (_Float16)bw1[k * 32 + n];
    Gw2[idx] = (n < out_dim) ? (_Float16)gw2[k * out_dim + n] : (_Float16)0.f;
    Bw2[idx] = (n < out_dim) ? (_Float16)bw2[k * out_dim + n] : (_Float16)0.f;
  }

  int tile = blockIdx.x * WAVES + wave;
  int n0 = tile * 16;
  _Float16* sta = &stage[wave][0];
  int col = lane & 15, hsel = lane >> 4;

  { // stage U = [h, agg, temb] zero-padded, fragment order, paired writes
    int row = lane >> 1, part = lane & 1;
    int n = n0 + row;
    bool v = n < Nn;
    float c = v ? cnt[n] : 1.f;
    float inv = 1.f / (c < 1.f ? 1.f : c);
    auto getf = [&](int k) -> float {
      if (!v) return 0.f;
      if (k < in_dim)      return h[(long)n * 32 + k];
      if (k < in_dim + 32) return s[(long)n * 32 + (k - in_dim)] * inv;
      if (k < in_dim + 34) return temb[(long)n * 2 + (k - in_dim - 32)];
      return 0.f;
    };
    int halfK = Kp2 >> 1;
    for (int kk = 0; kk < halfK; kk += 2) {
      int k = part * halfK + kk;
      v2h pr;
      pr[0] = (_Float16)getf(k);
      pr[1] = (_Float16)getf(k + 1);
      *(v2h*)(sta + (k >> 5) * 512 + frag_pos(row, k & 31)) = pr;
    }
  }
  __syncthreads();

  // phi GEMM1: (16 x Kp2) @ (Kp2 x 64)
  v8f acc[4] = {{}, {}, {}, {}};
  for (int wnd = 0; wnd < nw; ++wnd) {
    v16h a = load_frag(sta, wnd, lane);
#pragma unroll
    for (int nb = 0; nb < 4; ++nb) {
      v16h bf = load_frag(Pw1, wnd * 4 + nb, lane);
      acc[nb] = wmma16(a, bf, acc[nb]);
    }
  }
  __syncthreads();
  { // relu + bias -> hidden (16x64) in fragment order (windows 0..1)
#pragma unroll
    for (int j = 0; j < 8; ++j) {
      int M = j + hsel * 8;
#pragma unroll
      for (int nb = 0; nb < 4; ++nb) {
        int n = nb * 16 + col;
        float bias = (n < H2) ? pb1[n] : 0.f;
        float vv = acc[nb][j] + bias;
        sta[(n >> 5) * 512 + frag_pos(M, n & 31)] = (_Float16)(vv > 0.f ? vv : 0.f);
      }
    }
  }
  __syncthreads();

  // phi GEMM2: (16x64) @ (64 x outp)
  v8f phir[2] = {{}, {}};
  for (int wnd = 0; wnd < 2; ++wnd) {
    v16h a = load_frag(sta, wnd, lane);
#pragma unroll
    for (int nb = 0; nb < 2; ++nb)
      if (nb < noutb) {
        v16h bf = load_frag(Pw2, wnd * 2 + nb, lane);
        phir[nb] = wmma16(a, bf, phir[nb]);
      }
  }
  __syncthreads();

  // ---- gamma chain ----
  { // stage cond tile (window 0)
    int row = lane >> 1, part = lane & 1;
    int n = n0 + row;
    bool v = n < Nn;
    for (int kk = 0; kk < 16; kk += 2) {
      int k = part * 16 + kk;
      v2h pr;
      pr[0] = (_Float16)(v ? cemb[(long)n * 32 + k] : 0.f);
      pr[1] = (_Float16)(v ? cemb[(long)n * 32 + k + 1] : 0.f);
      *(v2h*)(sta + frag_pos(row, k)) = pr;
    }
  }
  __syncthreads();
  v8f g1[2] = {{}, {}};
  {
    v16h a = load_frag(sta, 0, lane);
#pragma unroll
    for (int nb = 0; nb < 2; ++nb) {
      v16h bf = load_frag(Gw1, nb, lane);
      g1[nb] = wmma16(a, bf, g1[nb]);
    }
  }
  __syncthreads();
  {
#pragma unroll
    for (int j = 0; j < 8; ++j) {
      int M = j + hsel * 8;
#pragma unroll
      for (int nb = 0; nb < 2; ++nb) {
        int n = nb * 16 + col;
        float vv = g1[nb][j] + gb1[n];
        sta[frag_pos(M, n)] = (_Float16)(vv > 0.f ? vv : 0.f);
      }
    }
  }
  __syncthreads();
  v8f gacc[2] = {{}, {}};
  {
    v16h a = load_frag(sta, 0, lane);
#pragma unroll
    for (int nb = 0; nb < 2; ++nb)
      if (nb < noutb) {
        v16h bf = load_frag(Gw2, nb, lane);
        gacc[nb] = wmma16(a, bf, gacc[nb]);
      }
  }
  __syncthreads();

  // ---- beta chain (restage cond; stage was clobbered) ----
  {
    int row = lane >> 1, part = lane & 1;
    int n = n0 + row;
    bool v = n < Nn;
    for (int kk = 0; kk < 16; kk += 2) {
      int k = part * 16 + kk;
      v2h pr;
      pr[0] = (_Float16)(v ? cemb[(long)n * 32 + k] : 0.f);
      pr[1] = (_Float16)(v ? cemb[(long)n * 32 + k + 1] : 0.f);
      *(v2h*)(sta + frag_pos(row, k)) = pr;
    }
  }
  __syncthreads();
  v8f be1[2] = {{}, {}};
  {
    v16h a = load_frag(sta, 0, lane);
#pragma unroll
    for (int nb = 0; nb < 2; ++nb) {
      v16h bf = load_frag(Bw1, nb, lane);
      be1[nb] = wmma16(a, bf, be1[nb]);
    }
  }
  __syncthreads();
  {
#pragma unroll
    for (int j = 0; j < 8; ++j) {
      int M = j + hsel * 8;
#pragma unroll
      for (int nb = 0; nb < 2; ++nb) {
        int n = nb * 16 + col;
        float vv = be1[nb][j] + bb1[n];
        sta[frag_pos(M, n)] = (_Float16)(vv > 0.f ? vv : 0.f);
      }
    }
  }
  __syncthreads();
  v8f bacc[2] = {{}, {}};
  {
    v16h a = load_frag(sta, 0, lane);
#pragma unroll
    for (int nb = 0; nb < 2; ++nb)
      if (nb < noutb) {
        v16h bf = load_frag(Bw2, nb, lane);
        bacc[nb] = wmma16(a, bf, bacc[nb]);
      }
  }

  // fused FiLM store: out = gamma * phi + beta
  {
#pragma unroll
    for (int nb = 0; nb < 2; ++nb)
      if (nb < noutb) {
#pragma unroll
        for (int j = 0; j < 8; ++j) {
          int M = j + hsel * 8;
          int n = n0 + M;
          int nn = nb * 16 + col;
          if (n < Nn && nn < out_dim) {
            float pv = phir[nb][j] + pb2[nn];
            float gv = gacc[nb][j] + gb2[nn];
            float bv = bacc[nb][j] + bb2[nn];
            out[(long)n * out_stride + nn] = gv * pv + bv;
          }
        }
      }
  }
}

extern "C" void kernel_launch(void* const* d_in, const int* in_sizes, int n_in,
                              void* d_out, int out_size, void* d_ws, size_t ws_size,
                              hipStream_t stream) {
  const float* x    = (const float*)d_in[0];
  const float* act  = (const float*)d_in[1];
  const int*   ei   = (const int*)d_in[2];
  const float* ea   = (const float*)d_in[3];
  const float* y    = (const float*)d_in[4];
  const int*   bidx = (const int*)d_in[5];
  const float* tim  = (const float*)d_in[6];
  const float* cw1  = (const float*)d_in[7];
  const float* cb1  = (const float*)d_in[8];
  const float* cw2  = (const float*)d_in[9];
  const float* cb2  = (const float*)d_in[10];

  const int Nn = in_sizes[5];
  const int En = in_sizes[2] / 2;

  float* hA   = (float*)d_ws;
  float* hB   = hA + (size_t)Nn * 32;
  float* sbuf = hB + (size_t)Nn * 32;
  float* cnt  = sbuf + (size_t)Nn * 32;
  float* temb = cnt + (size_t)Nn;
  float* cemb = temb + (size_t)Nn * 2;

  zero_kernel<<<(Nn + 255) / 256, 256, 0, stream>>>(cnt, Nn);
  count_kernel<<<(En + 255) / 256, 256, 0, stream>>>(ei, cnt, En);
  prep_kernel<<<(Nn + 255) / 256, 256, 0, stream>>>(x, act, bidx, tim, y,
                                                    cw1, cb1, cw2, cb2,
                                                    hA, temb, cemb, Nn);

  const int in_dims[4]  = {12, 32, 32, 32};
  const int out_dims[4] = {32, 32, 32, 10};
  float* hcur = hA;
  float* hnext = hB;
  const int tilesE = (En + 15) / 16;
  const int tilesN = (Nn + 15) / 16;

  for (int l = 0; l < 4; ++l) {
    // params flattened (jax tree order): cond, then per-layer keys sorted:
    // beta(w1,b1,w2,b2), gamma(...), phi(...), psi(...)
    int base = 11 + 16 * l;
    const float* bw1_ = (const float*)d_in[base + 0];
    const float* bb1_ = (const float*)d_in[base + 1];
    const float* bw2_ = (const float*)d_in[base + 2];
    const float* bb2_ = (const float*)d_in[base + 3];
    const float* gw1_ = (const float*)d_in[base + 4];
    const float* gb1_ = (const float*)d_in[base + 5];
    const float* gw2_ = (const float*)d_in[base + 6];
    const float* gb2_ = (const float*)d_in[base + 7];
    const float* pw1_ = (const float*)d_in[base + 8];
    const float* pb1_ = (const float*)d_in[base + 9];
    const float* pw2_ = (const float*)d_in[base + 10];
    const float* pb2_ = (const float*)d_in[base + 11];
    const float* sw1_ = (const float*)d_in[base + 12];
    const float* sb1_ = (const float*)d_in[base + 13];
    const float* sw2_ = (const float*)d_in[base + 14];
    const float* sb2_ = (const float*)d_in[base + 15];

    int ind = in_dims[l], outd = out_dims[l];

    zero_kernel<<<((Nn * 32) + 255) / 256, 256, 0, stream>>>(sbuf, Nn * 32);
    edge_kernel<<<(tilesE + WAVES - 1) / WAVES, 256, 0, stream>>>(
        hcur, ind, ei, ea, sw1_, sb1_, sw2_, sb2_, sbuf, En);

    float* outp = (l == 3) ? (float*)d_out : hnext;
    int ostride = (l == 3) ? 10 : 32;
    node_kernel<<<(tilesN + WAVES - 1) / WAVES, 256, 0, stream>>>(
        hcur, ind, outd, sbuf, cnt, temb, cemb,
        pw1_, pb1_, pw2_, pb2_,
        gw1_, gb1_, gw2_, gb2_,
        bw1_, bb1_, bw2_, bb2_,
        outp, ostride, Nn);

    float* tmp = hcur; hcur = hnext; hnext = tmp;
  }
}
